// EnergyAttention_15582141350429
// MI455X (gfx1250) — matrix operand order repaired and checked by reading, not compiled, not run on hardware
//
#include <hip/hip_runtime.h>
#include <hip/hip_bf16.h>

typedef __attribute__((ext_vector_type(16))) __bf16 v16bf;
typedef __attribute__((ext_vector_type(8)))  float  v8f;

#define N_TOK 2048
#define D_DIM 768
#define H_NUM 12
#define Y_DIM 64
#define HY    (H_NUM * Y_DIM)    // 768
#define KTILES (N_TOK / 16)      // 128
#define QPAIRS (N_TOK / 32)      // 64  (two 16-row q-tiles per wave)
#define NWAVES2 (H_NUM * QPAIRS) // 768 waves in stage 2

union Frag16 { v16bf v; uint4 q[2]; unsigned int u[8]; };

// log2(e) * beta  and  ln2 for the final unscale
#define SCALE_LOG2E_BETA 0.18033688011112042f   // 0.125 * 1.4426950408889634
#define NEG_SQRT_Y_LN2  -5.545177444479562f     // -8 * ln(2)

// ---------------------------------------------------------------------------
// Stage 1: Q = g @ Wq^T, K = g @ Wk^T   (f32 in, bf16 out, f32 accumulate)
// One wave -> one 16x16 tile. Fragment runs are contiguous -> b128 loads.
//   A (16x32): lane row = lane&15; elems [0..7] = K 8g..8g+7, [8..15] = K 16+8g..
//   B (32x16): lane col = lane&15; elems [0..7] = K 8g..8g+7, [8..15] = K 16+8g..
// ---------------------------------------------------------------------------
__global__ __launch_bounds__(256) void proj_kernel(
    const float* __restrict__ g, const float* __restrict__ Wq,
    const float* __restrict__ Wk, __bf16* __restrict__ Qb,
    __bf16* __restrict__ Kb)
{
  const float* W = (blockIdx.z == 0) ? Wq : Wk;
  __bf16* out    = (blockIdx.z == 0) ? Qb : Kb;

  const int tid  = threadIdx.x;
  const int wave = tid >> 5;
  const int lane = tid & 31;
  const int g16  = lane >> 4;
  const int ln   = lane & 15;

  const int rowBase = blockIdx.y * 16;              // token rows
  const int colBase = blockIdx.x * 128 + wave * 16; // output cols (= h*64+z)

  v8f c = {};
  for (int k0 = 0; k0 < D_DIM; k0 += 32) {
    const float* ga = g + (size_t)(rowBase + ln) * D_DIM + k0;
    const float* wa = W + (size_t)(colBase + ln) * D_DIM + k0;

    // A: two contiguous 8-float runs -> 4x float4 (global_load_b128)
    const float4 fa0 = *(const float4*)(ga + 8 * g16);
    const float4 fa1 = *(const float4*)(ga + 8 * g16 + 4);
    const float4 fa2 = *(const float4*)(ga + 16 + 8 * g16);
    const float4 fa3 = *(const float4*)(ga + 16 + 8 * g16 + 4);
    // B: two contiguous 8-float runs -> 4x float4
    const float4 fb0 = *(const float4*)(wa + 8 * g16);
    const float4 fb1 = *(const float4*)(wa + 8 * g16 + 4);
    const float4 fb2 = *(const float4*)(wa + 16 + 8 * g16);
    const float4 fb3 = *(const float4*)(wa + 16 + 8 * g16 + 4);

    v16bf af, bf;
    af[0]=(__bf16)fa0.x; af[1]=(__bf16)fa0.y; af[2]=(__bf16)fa0.z; af[3]=(__bf16)fa0.w;
    af[4]=(__bf16)fa1.x; af[5]=(__bf16)fa1.y; af[6]=(__bf16)fa1.z; af[7]=(__bf16)fa1.w;
    af[8]=(__bf16)fa2.x; af[9]=(__bf16)fa2.y; af[10]=(__bf16)fa2.z; af[11]=(__bf16)fa2.w;
    af[12]=(__bf16)fa3.x; af[13]=(__bf16)fa3.y; af[14]=(__bf16)fa3.z; af[15]=(__bf16)fa3.w;
    bf[0]=(__bf16)fb0.x; bf[1]=(__bf16)fb0.y; bf[2]=(__bf16)fb0.z; bf[3]=(__bf16)fb0.w;
    bf[4]=(__bf16)fb1.x; bf[5]=(__bf16)fb1.y; bf[6]=(__bf16)fb1.z; bf[7]=(__bf16)fb1.w;
    bf[8]=(__bf16)fb2.x; bf[9]=(__bf16)fb2.y; bf[10]=(__bf16)fb2.z; bf[11]=(__bf16)fb2.w;
    bf[12]=(__bf16)fb3.x; bf[13]=(__bf16)fb3.y; bf[14]=(__bf16)fb3.z; bf[15]=(__bf16)fb3.w;

    c = __builtin_amdgcn_wmma_f32_16x16x32_bf16(false, af, false, bf,
                                                (short)0, c, false, false);
  }
  // C layout: lane holds col n = ln, rows r + 8*g16 in VGPR r
#pragma unroll
  for (int r = 0; r < 8; ++r) {
    const int row = rowBase + r + 8 * g16;
    out[(size_t)row * HY + colBase + ln] = (__bf16)c[r];
  }
}

// ---------------------------------------------------------------------------
// Stage 2: fused scores + online logsumexp (base-2). One wave handles one
// head and TWO 16-row q-tiles, reusing each K-tile B-fragment twice
// (halves L2 read bandwidth). Score tensor never materialized.
// ---------------------------------------------------------------------------
__global__ __launch_bounds__(256) void attn_lse_kernel(
    const __bf16* __restrict__ Qb, const __bf16* __restrict__ Kb,
    float* __restrict__ partials)
{
  const int tid  = threadIdx.x;
  const int wid  = blockIdx.x * 8 + (tid >> 5);   // 0..767
  const int lane = tid & 31;
  const int g16  = lane >> 4;
  const int ln   = lane & 15;
  const int h    = wid / QPAIRS;
  const int q0   = (wid % QPAIRS) * 32;

  // Q A-fragments: two q-tiles x two z-halves, b128 loads, loaded once
  Frag16 a[2][2];
#pragma unroll
  for (int t = 0; t < 2; ++t) {
    const __bf16* qrow = Qb + (size_t)(q0 + t * 16 + ln) * HY + h * Y_DIM;
    a[t][0].q[0] = *(const uint4*)(qrow + 8 * g16);
    a[t][0].q[1] = *(const uint4*)(qrow + 16 + 8 * g16);
    a[t][1].q[0] = *(const uint4*)(qrow + 32 + 8 * g16);
    a[t][1].q[1] = *(const uint4*)(qrow + 48 + 8 * g16);
  }

  float lm[2][8], ls[2][8];
#pragma unroll
  for (int t = 0; t < 2; ++t)
#pragma unroll
    for (int r = 0; r < 8; ++r) { lm[t][r] = -__builtin_inff(); ls[t][r] = 0.0f; }

  for (int kt = 0; kt < KTILES; ++kt) {
    const __bf16* krow = Kb + (size_t)(kt * 16 + ln) * HY + h * Y_DIM;
    Frag16 b0, b1;                       // B: two 8-elem runs (8g, 16+8g) -> b128
    b0.q[0] = *(const uint4*)(krow + 8 * g16);
    b0.q[1] = *(const uint4*)(krow + 16 + 8 * g16);
    b1.q[0] = *(const uint4*)(krow + 32 + 8 * g16);
    b1.q[1] = *(const uint4*)(krow + 48 + 8 * g16);
    if (kt + 1 < KTILES)                 // global_prefetch_b8 next key tile
      __builtin_prefetch(Kb + (size_t)((kt + 1) * 16 + ln) * HY + h * Y_DIM, 0, 3);

#pragma unroll
    for (int t = 0; t < 2; ++t) {
      v8f c = {};
      c = __builtin_amdgcn_wmma_f32_16x16x32_bf16(false, a[t][0].v, false, b0.v,
                                                  (short)0, c, false, false);
      c = __builtin_amdgcn_wmma_f32_16x16x32_bf16(false, a[t][1].v, false, b1.v,
                                                  (short)0, c, false, false);
      // base-2 online logsumexp: v_exp_f32 is natively 2^x, no extra mul
#pragma unroll
      for (int r = 0; r < 8; ++r) {
        const float v  = c[r] * SCALE_LOG2E_BETA;
        const float mN = fmaxf(lm[t][r], v);
        ls[t][r] = ls[t][r] * __builtin_amdgcn_exp2f(lm[t][r] - mN)
                 + __builtin_amdgcn_exp2f(v - mN);
        lm[t][r] = mN;
      }
    }
  }

  // merge (m,s) across the 16 lanes of each half-wave, then sum all 32 rows
  float acc = 0.0f;
#pragma unroll
  for (int t = 0; t < 2; ++t) {
#pragma unroll
    for (int r = 0; r < 8; ++r) {
      float m = lm[t][r], s = ls[t][r];
#pragma unroll
      for (int off = 1; off < 16; off <<= 1) {
        const float mo = __shfl_xor(m, off, 16);
        const float so = __shfl_xor(s, off, 16);
        const float mN = fmaxf(m, mo);
        s = s * __builtin_amdgcn_exp2f(m - mN) + so * __builtin_amdgcn_exp2f(mo - mN);
        m = mN;
      }
      acc += m + __builtin_amdgcn_logf(s);  // log2-domain lse for this row
    }
  }
  acc += __shfl_xor(acc, 16, 32);  // combine the two half-wave row groups
  if (lane == 0) partials[wid] = acc;
}

// ---------------------------------------------------------------------------
// Stage 3: deterministic fixed-order reduction; convert log2 -> ln here
// ---------------------------------------------------------------------------
__global__ __launch_bounds__(256) void reduce_kernel(
    const float* __restrict__ partials, float* __restrict__ out)
{
  __shared__ float sm[256];
  float s = 0.0f;
  for (int i = threadIdx.x; i < NWAVES2; i += 256) s += partials[i];
  sm[threadIdx.x] = s;
  __syncthreads();
  for (int st = 128; st > 0; st >>= 1) {
    if ((int)threadIdx.x < st) sm[threadIdx.x] += sm[threadIdx.x + st];
    __syncthreads();
  }
  if (threadIdx.x == 0) out[0] = NEG_SQRT_Y_LN2 * sm[0];  // (-1/beta)*ln2*sum
}

extern "C" void kernel_launch(void* const* d_in, const int* in_sizes, int n_in,
                              void* d_out, int out_size, void* d_ws, size_t ws_size,
                              hipStream_t stream) {
  (void)in_sizes; (void)n_in; (void)out_size; (void)ws_size;
  const float* g  = (const float*)d_in[0];
  const float* Wq = (const float*)d_in[1];
  const float* Wk = (const float*)d_in[2];

  __bf16* Qb = (__bf16*)d_ws;                          // [N, HY] bf16
  __bf16* Kb = Qb + (size_t)N_TOK * HY;                // [N, HY] bf16
  float* partials = (float*)(Kb + (size_t)N_TOK * HY); // [768] f32

  dim3 pgrid(HY / 128, N_TOK / 16, 2);  // (6, 128, 2)
  proj_kernel<<<pgrid, 256, 0, stream>>>(g, Wq, Wk, Qb, Kb);

  attn_lse_kernel<<<dim3(NWAVES2 / 8), 256, 0, stream>>>(Qb, Kb, partials);

  reduce_kernel<<<dim3(1), 256, 0, stream>>>(partials, (float*)d_out);
}
